// Attention_11948599017728
// MI455X (gfx1250) — matrix operand order repaired and checked by reading, not compiled, not run on hardware
//
#include <hip/hip_runtime.h>
#include <hip/hip_bf16.h>

typedef __attribute__((ext_vector_type(16))) _Float16 v16h;
typedef __attribute__((ext_vector_type(8)))  _Float16 v8h;
typedef __attribute__((ext_vector_type(8)))  float    v8f;
typedef int v4i __attribute__((vector_size(16)));
typedef __attribute__((address_space(1))) v4i* gas_v4i;
typedef __attribute__((address_space(3))) v4i* las_v4i;

#define DIM   384
#define HEADS 8
#define HD    48
#define HDP   64            // hd padded to 64 (two K=32 WMMA chunks)
#define BATCH 4
#define NTOK  4097
#define MTOK  513
#define MPAD  544           // 17 * 32
#define NPAD  4160          // 65 * 64
#define BH    (BATCH*HEADS)
#define QK_SCALE 0.14433756729740643f   // 48^-0.5

#if __has_builtin(__builtin_amdgcn_global_load_async_to_lds_b128)
#define HAS_ASYNC_LDS 1
#endif

// Copy 16B global -> LDS. Async (ASYNCcnt-tracked, no VGPR round-trip) when available.
__device__ __forceinline__ void cp_b128(_Float16* lds, const _Float16* g) {
#ifdef HAS_ASYNC_LDS
  __builtin_amdgcn_global_load_async_to_lds_b128(
      (gas_v4i)(void*)g, (las_v4i)(void*)lds, 0, 0);
#else
  *(v8h*)lds = *(const v8h*)g;
#endif
}

template <int N>
__device__ __forceinline__ void async_wait() {
#ifdef HAS_ASYNC_LDS
#if __has_builtin(__builtin_amdgcn_s_wait_asynccnt)
  __builtin_amdgcn_s_wait_asynccnt(N);
#else
  asm volatile("s_wait_asynccnt %0" :: "i"(N) : "memory");
#endif
#endif
}

// Build a v16h WMMA fragment from two 16-byte chunks (LDS or global).
__device__ __forceinline__ v16h ld_frag(const _Float16* p0, const _Float16* p1) {
  v8h lo = *(const v8h*)p0;
  v8h hi = *(const v8h*)p1;
  v16h r;
#pragma unroll
  for (int i = 0; i < 8; ++i) { r[i] = lo[i]; r[i + 8] = hi[i]; }
  return r;
}

__device__ __forceinline__ v8f wmma_f16(v16h a, v16h b, v8f c) {
  return __builtin_amdgcn_wmma_f32_16x16x32_f16(false, a, false, b, (short)0, c,
                                                false, false);
}

// ---------------------------------------------------------------- converts
__global__ void k_f32_to_f16(const float* __restrict__ s, _Float16* __restrict__ d, int n) {
  int i = blockIdx.x * 256 + threadIdx.x;
  if (i < n) d[i] = (_Float16)s[i];
}

// ------------------------------------------- spatial reduction + LayerNorm
// grid = B*M blocks, 128 threads; each thread owns 3 channels.
__global__ void __launch_bounds__(128)
k_sr_ln(const float* __restrict__ x, const float* __restrict__ srw,
        const float* __restrict__ srb, const float* __restrict__ lng,
        const float* __restrict__ lnb, _Float16* __restrict__ xlnh) {
  const int bm = blockIdx.x;
  const int b = bm / MTOK, m = bm % MTOK;
  const int tid = threadIdx.x;
  float vals[3];
  if (m == 0) {
#pragma unroll
    for (int j = 0; j < 3; ++j)
      vals[j] = x[(size_t)b * NTOK * DIM + tid + j * 128];
  } else {
    int s = m - 1;
    int o0 = s >> 6, o1 = (s >> 3) & 7, o2 = s & 7;
    int i0 = o0 * 2, i1 = o1 * 2, i2 = o2 * 2;
#pragma unroll
    for (int j = 0; j < 3; ++j) {
      int c = tid + j * 128;
      float acc = srb[c];
#pragma unroll
      for (int t = 0; t < 8; ++t) {
        int dz = t >> 2, dy = (t >> 1) & 1, dx = t & 1;
        int n = 1 + ((i0 + dz) * 16 + (i1 + dy)) * 16 + (i2 + dx);
        acc += x[((size_t)b * NTOK + n) * DIM + c] * srw[c * 8 + t];
      }
      vals[j] = acc;
    }
  }
  __shared__ float s1[128], s2[128];
  float a = 0.f, q = 0.f;
#pragma unroll
  for (int j = 0; j < 3; ++j) { a += vals[j]; q += vals[j] * vals[j]; }
  s1[tid] = a; s2[tid] = q;
  __syncthreads();
  for (int st = 64; st > 0; st >>= 1) {
    if (tid < st) { s1[tid] += s1[tid + st]; s2[tid] += s2[tid + st]; }
    __syncthreads();
  }
  float mean = s1[0] * (1.0f / DIM);
  float var  = s2[0] * (1.0f / DIM) - mean * mean;
  float rstd = rsqrtf(var + 1e-5f);
#pragma unroll
  for (int j = 0; j < 3; ++j) {
    int c = tid + j * 128;
    xlnh[(size_t)bm * DIM + c] = (_Float16)((vals[j] - mean) * rstd * lng[c] + lnb[c]);
  }
}

// ---------------------------------------------------------------- WMMA GEMM
// C[rows,Nout] = A[rows,384] * W[Nout,384]^T (+bias).
// 256 threads = 8 waves (4 row x 2 col); block tile 128x64; wave tile 32x32
// (2x2 WMMA frags). Double-buffered async global->LDS staging.
__global__ void __launch_bounds__(256)
k_gemm_f16(const _Float16* __restrict__ A, const _Float16* __restrict__ W,
           const float* __restrict__ bias, float* __restrict__ C,
           int rows, int Nout) {
  __shared__ __align__(16) _Float16 As[2][128 * 32];
  __shared__ __align__(16) _Float16 Ws[2][64 * 32];
  const int tid  = threadIdx.x;
  const int wave = tid >> 5, lane = tid & 31;
  const int rw = wave >> 1, cw = wave & 1;
  const int lr = lane & 15, hi = lane >> 4;
  const int row0 = blockIdx.x * 128;
  const int col0 = blockIdx.y * 64;

  auto stage = [&](int bb, int k0) {
    {   // A tile: 128 rows x 32 halves = 8 KB, 2 x b128 per thread
      int r = tid >> 1, s = (tid & 1) * 16;
      int gr = row0 + r; if (gr > rows - 1) gr = rows - 1;
      const _Float16* src = A + (size_t)gr * DIM + k0 + s;
      cp_b128(&As[bb][r * 32 + s], src);
      cp_b128(&As[bb][r * 32 + s + 8], src + 8);
    }
    {   // W tile: 64 cols x 32 halves = 4 KB, 1 x b128 per thread
      int r = tid >> 2, s = (tid & 3) * 8;
      cp_b128(&Ws[bb][r * 32 + s], W + (size_t)(col0 + r) * DIM + k0 + s);
    }
  };

  stage(0, 0);
  v8f acc00 = {}, acc01 = {}, acc10 = {}, acc11 = {};
  int buf = 0;
#pragma unroll
  for (int k0 = 0; k0 < DIM; k0 += 32) {
    if (k0 + 32 < DIM) { stage(buf ^ 1, k0 + 32); async_wait<3>(); }
    else               { async_wait<0>(); }
    __syncthreads();
    const _Float16* ab = &As[buf][rw * 32 * 32];
    v16h a0 = ld_frag(ab + lr * 32 + hi * 8,        ab + lr * 32 + 16 + hi * 8);
    v16h a1 = ld_frag(ab + (16 + lr) * 32 + hi * 8, ab + (16 + lr) * 32 + 16 + hi * 8);
    const _Float16* wb = &Ws[buf][cw * 32 * 32];
    v16h b0 = ld_frag(wb + lr * 32 + hi * 8,        wb + lr * 32 + 16 + hi * 8);
    v16h b1 = ld_frag(wb + (16 + lr) * 32 + hi * 8, wb + (16 + lr) * 32 + 16 + hi * 8);
    acc00 = wmma_f16(a0, b0, acc00);
    acc01 = wmma_f16(a0, b1, acc01);
    acc10 = wmma_f16(a1, b0, acc10);
    acc11 = wmma_f16(a1, b1, acc11);
    __syncthreads();
    buf ^= 1;
  }

#define STORE_TILE(ACC, RI, CI)                                            \
  do {                                                                     \
    int cc = col0 + cw * 32 + (CI) * 16 + lr;                              \
    float bv = bias ? bias[cc] : 0.0f;                                     \
    _Pragma("unroll") for (int v = 0; v < 8; ++v) {                        \
      int rr = row0 + rw * 32 + (RI) * 16 + hi * 8 + v;                    \
      if (rr < rows) C[(size_t)rr * Nout + cc] = ACC[v] + bv;              \
    }                                                                      \
  } while (0)
  STORE_TILE(acc00, 0, 0);
  STORE_TILE(acc01, 0, 1);
  STORE_TILE(acc10, 1, 0);
  STORE_TILE(acc11, 1, 1);
#undef STORE_TILE
}

// -------------------------------------------------------- scatter / repack
__global__ void k_q_scatter(const float* __restrict__ qf, _Float16* __restrict__ qh) {
  int i = blockIdx.x * 256 + threadIdx.x;
  if (i >= BH * NPAD * HDP) return;
  int d  = i & 63;
  int n  = (i >> 6) % NPAD;
  int bh = i / (NPAD * HDP);
  int b = bh >> 3, h = bh & 7;
  float val = 0.0f;
  if (n < NTOK && d < HD)
    val = qf[((size_t)b * NTOK + n) * DIM + h * HD + d] * QK_SCALE;
  qh[i] = (_Float16)val;
}

__global__ void k_kv_scatter(const float* __restrict__ kvf, _Float16* __restrict__ kh,
                             _Float16* __restrict__ vth) {
  int i = blockIdx.x * 256 + threadIdx.x;
  if (i >= BH * MPAD * HDP) return;
  int d  = i & 63;
  int m  = (i >> 6) % MPAD;
  int bh = i / (MPAD * HDP);
  int b = bh >> 3, h = bh & 7;
  float kv = 0.0f, vv = 0.0f;
  if (m < MTOK && d < HD) {
    size_t base = ((size_t)b * MTOK + m) * (2 * DIM) + h * HD + d;
    kv = kvf[base];
    vv = kvf[base + DIM];
  }
  kh[i] = (_Float16)kv;                                    // [bh][m][64]
  vth[((size_t)bh * HDP + d) * MPAD + m] = (_Float16)vv;   // [bh][64][m]
}

// -------------------------------------------------- flash attention (WMMA)
// grid = (NPAD/64, B*H); 128 threads = 4 waves, 16 q-rows per wave.
// Double-buffered async K/V tile staging.
__global__ void __launch_bounds__(128)
k_attn(const _Float16* __restrict__ qh, const _Float16* __restrict__ kh,
       const _Float16* __restrict__ vth, _Float16* __restrict__ outh) {
  __shared__ __align__(16) _Float16 kt[2][32 * 64];     // [m][d]
  __shared__ __align__(16) _Float16 vt[2][64 * 32];     // [d][m]
  __shared__ __align__(16) _Float16 pt[4 * 16 * 32];    // per-wave P staging
  const int bh = blockIdx.y;
  const int b = bh >> 3, h = bh & 7;
  const int tid = threadIdx.x;
  const int wave = tid >> 5, lane = tid & 31;
  const int lr = lane & 15, hi = lane >> 4;
  const int n0 = blockIdx.x * 64 + wave * 16;

  // Q A-fragments held in registers for the whole M loop (scale pre-folded).
  const _Float16* qrow = qh + ((size_t)bh * NPAD + n0 + lr) * HDP;
  v16h qa0 = ld_frag(qrow + hi * 16,      qrow + hi * 16 + 8);
  v16h qa1 = ld_frag(qrow + 32 + hi * 16, qrow + 32 + hi * 16 + 8);

  float mx[8], ls[8];
#pragma unroll
  for (int v = 0; v < 8; ++v) { mx[v] = -1e30f; ls[v] = 0.0f; }
  v8f o0 = {}, o1 = {}, o2 = {};

  const _Float16* kbh = kh  + (size_t)bh * MPAD * HDP;
  const _Float16* vbh = vth + (size_t)bh * HDP * MPAD;
  _Float16* pw = &pt[wave * (16 * 32)];

  auto stage = [&](int bb, int mt_) {
    {   // K tile [32 m][64 d]
      int r = tid >> 2, s = (tid & 3) * 16;
      const _Float16* src = kbh + (size_t)(mt_ * 32 + r) * HDP + s;
      cp_b128(&kt[bb][r * 64 + s], src);
      cp_b128(&kt[bb][r * 64 + s + 8], src + 8);
    }
    {   // V^T tile [64 d][32 m]
      int r = tid >> 1, s = (tid & 1) * 16;
      const _Float16* src = vbh + (size_t)r * MPAD + mt_ * 32 + s;
      cp_b128(&vt[bb][r * 32 + s], src);
      cp_b128(&vt[bb][r * 32 + s + 8], src + 8);
    }
  };

  stage(0, 0);
  int buf = 0;
  for (int mt = 0; mt < MPAD / 32; ++mt) {
    if (mt + 1 < MPAD / 32) { stage(buf ^ 1, mt + 1); async_wait<4>(); }
    else                    { async_wait<0>(); }
    __syncthreads();
    const _Float16* ktb = kt[buf];
    const _Float16* vtb = vt[buf];

    // S = q . k^T for two 16-wide m-subtiles (B frag: col=m, K=d contiguous)
    v8f s0 = {}, s1 = {};
    {
      const _Float16* kp = &ktb[lr * 64];
      s0 = wmma_f16(qa0, ld_frag(kp + hi * 16,      kp + hi * 16 + 8),      s0);
      s0 = wmma_f16(qa1, ld_frag(kp + 32 + hi * 16, kp + 32 + hi * 16 + 8), s0);
    }
    {
      const _Float16* kp = &ktb[(16 + lr) * 64];
      s1 = wmma_f16(qa0, ld_frag(kp + hi * 16,      kp + hi * 16 + 8),      s1);
      s1 = wmma_f16(qa1, ld_frag(kp + 32 + hi * 16, kp + 32 + hi * 16 + 8), s1);
    }

    const bool ok0 = (mt * 32 + lr)      < MTOK;
    const bool ok1 = (mt * 32 + 16 + lr) < MTOK;
#pragma unroll
    for (int v = 0; v < 8; ++v) {
      float a = ok0 ? s0[v] : -1e30f;
      float c = ok1 ? s1[v] : -1e30f;
      float tm = fmaxf(a, c);
#pragma unroll
      for (int off = 8; off; off >>= 1) tm = fmaxf(tm, __shfl_xor(tm, off, 32));
      float nm = fmaxf(mx[v], tm);
      float corr = __expf(mx[v] - nm);
      mx[v] = nm;
      float e0 = ok0 ? __expf(a - nm) : 0.0f;
      float e1 = ok1 ? __expf(c - nm) : 0.0f;
      float rs = e0 + e1;
#pragma unroll
      for (int off = 8; off; off >>= 1) rs += __shfl_xor(rs, off, 32);
      ls[v] = ls[v] * corr + rs;
      o0[v] *= corr; o1[v] *= corr; o2[v] *= corr;
      // stash P in row-major [16][32] to re-enter as an A fragment
      pw[(hi * 8 + v) * 32 + lr]      = (_Float16)e0;
      pw[(hi * 8 + v) * 32 + 16 + lr] = (_Float16)e1;
    }
    asm volatile("s_wait_dscnt 0" ::: "memory");  // same-wave DS RAW fence

    v16h pa = ld_frag(pw + lr * 32 + hi * 16, pw + lr * 32 + hi * 16 + 8);
    {  // O += P . V  (B frag: col=d, K=m contiguous since V stored transposed)
      const _Float16* vp = &vtb[lr * 32 + hi * 16];
      o0 = wmma_f16(pa, ld_frag(vp, vp + 8), o0);
    }
    {
      const _Float16* vp = &vtb[(16 + lr) * 32 + hi * 16];
      o1 = wmma_f16(pa, ld_frag(vp, vp + 8), o1);
    }
    {
      const _Float16* vp = &vtb[(32 + lr) * 32 + hi * 16];
      o2 = wmma_f16(pa, ld_frag(vp, vp + 8), o2);
    }
    __syncthreads();
    buf ^= 1;
  }

#pragma unroll
  for (int v = 0; v < 8; ++v) {
    int n = n0 + hi * 8 + v;
    if (n < NTOK) {
      float inv = 1.0f / ls[v];
      size_t base = ((size_t)b * NTOK + n) * DIM + h * HD;
      outh[base + lr]      = (_Float16)(o0[v] * inv);
      outh[base + 16 + lr] = (_Float16)(o1[v] * inv);
      outh[base + 32 + lr] = (_Float16)(o2[v] * inv);
    }
  }
}

// ------------------------------------------------------------------ launch
extern "C" void kernel_launch(void* const* d_in, const int* in_sizes, int n_in,
                              void* d_out, int out_size, void* d_ws, size_t ws_size,
                              hipStream_t stream) {
  const float* x      = (const float*)d_in[0];
  const float* q_w    = (const float*)d_in[1];
  const float* kv_w   = (const float*)d_in[2];
  const float* proj_w = (const float*)d_in[3];
  const float* proj_b = (const float*)d_in[4];
  const float* sr_w   = (const float*)d_in[5];
  const float* sr_b   = (const float*)d_in[6];
  const float* ln_g   = (const float*)d_in[7];
  const float* ln_b   = (const float*)d_in[8];

  char* ws = (char*)d_ws;
  size_t off = 0;
  auto alloc = [&](size_t bytes) -> char* {
    char* p = ws + off;
    off = (off + bytes + 255) & ~(size_t)255;
    return p;
  };
  const int rowsX = BATCH * NTOK;   // 16388
  const int rowsM = BATCH * MTOK;   // 2052
  _Float16* xh    = (_Float16*)alloc((size_t)rowsX * DIM * 2);
  _Float16* wq_h  = (_Float16*)alloc((size_t)DIM * DIM * 2);
  _Float16* wkv_h = (_Float16*)alloc((size_t)2 * DIM * DIM * 2);
  _Float16* wp_h  = (_Float16*)alloc((size_t)DIM * DIM * 2);
  _Float16* xln_h = (_Float16*)alloc((size_t)rowsM * DIM * 2);
  float*    qf    = (float*)   alloc((size_t)rowsX * DIM * 4);
  float*    kvf   = (float*)   alloc((size_t)rowsM * 2 * DIM * 4);
  _Float16* qh    = (_Float16*)alloc((size_t)BH * NPAD * HDP * 2);
  _Float16* kh    = (_Float16*)alloc((size_t)BH * MPAD * HDP * 2);
  _Float16* vth   = (_Float16*)alloc((size_t)BH * MPAD * HDP * 2);
  _Float16* aoh   = (_Float16*)alloc((size_t)rowsX * DIM * 2);

  int n;
  n = rowsX * DIM;
  k_f32_to_f16<<<(n + 255) / 256, 256, 0, stream>>>(x, xh, n);
  n = DIM * DIM;
  k_f32_to_f16<<<(n + 255) / 256, 256, 0, stream>>>(q_w, wq_h, n);
  n = 2 * DIM * DIM;
  k_f32_to_f16<<<(n + 255) / 256, 256, 0, stream>>>(kv_w, wkv_h, n);
  n = DIM * DIM;
  k_f32_to_f16<<<(n + 255) / 256, 256, 0, stream>>>(proj_w, wp_h, n);

  k_sr_ln<<<BATCH * MTOK, 128, 0, stream>>>(x, sr_w, sr_b, ln_g, ln_b, xln_h);

  k_gemm_f16<<<dim3((rowsX + 127) / 128, DIM / 64), 256, 0, stream>>>(
      xh, wq_h, nullptr, qf, rowsX, DIM);
  k_gemm_f16<<<dim3((rowsM + 127) / 128, (2 * DIM) / 64), 256, 0, stream>>>(
      xln_h, wkv_h, nullptr, kvf, rowsM, 2 * DIM);

  n = BH * NPAD * HDP;
  k_q_scatter<<<(n + 255) / 256, 256, 0, stream>>>(qf, qh);
  n = BH * MPAD * HDP;
  k_kv_scatter<<<(n + 255) / 256, 256, 0, stream>>>(kvf, kh, vth);

  k_attn<<<dim3(NPAD / 64, BH), 128, 0, stream>>>(qh, kh, vth, aoh);

  k_gemm_f16<<<dim3((rowsX + 127) / 128, DIM / 64), 256, 0, stream>>>(
      aoh, wp_h, proj_b, (float*)d_out, rowsX, DIM);

  (void)in_sizes; (void)n_in; (void)out_size; (void)ws_size;
}
